// TraceSetLSTM_16054587752962
// MI455X (gfx1250) — compile-verified, matrix operand order checked by reading, not run on hardware
//
#include <hip/hip_runtime.h>
#include <hip/hip_bf16.h>
#include <stdint.h>

typedef _Float16 h16;
typedef __attribute__((ext_vector_type(16))) _Float16 v16h;
typedef __attribute__((ext_vector_type(8)))  float    v8f;

#define DEV __device__ __forceinline__

DEV float sigf(float x) { return 1.0f / (1.0f + __expf(-x)); }

// ---------------------------------------------------------------------------
// WMMA helpers (CDNA5 v_wmma_f32_16x16x32_f16, wave32)
// Fragment layout per ISA 7.12.2 (16-bit A 16x32):
//   lanes 0-15 : row = lane,    elems 0..7 = K[0..7],  elems 8..15 = K[16..23]
//   lanes 16-31: row = lane-16, elems 0..7 = K[8..15], elems 8..15 = K[24..31]
// B operand uses the mirrored layout with rows = N (weights stored [N][K]).
// ---------------------------------------------------------------------------
DEV v16h load_frag(const h16* __restrict__ p_tile, int ld) {
  const int lane = threadIdx.x & 31;
  const int r    = lane & 15;
  const int kh   = (lane & 16) ? 8 : 0;
  const h16* p = p_tile + (size_t)r * ld + kh;
  v16h f;
#pragma unroll
  for (int i = 0; i < 8; ++i) { f[i] = p[i]; f[i + 8] = p[i + 16]; }
  return f;
}

DEV v8f wmma_f16(v16h a, v16h b, v8f c) {
  return __builtin_amdgcn_wmma_f32_16x16x32_f16(false, a, false, b, (short)0, c,
                                                false, false);
}

// ---------------------------------------------------------------------------
// Set encoder: one workgroup (128 thr / 4 waves) per (b,t,team).
// h1 = relu(pts@w1+b1) (11x128, VALU, K=2); h2 = relu(h1@w2+b2) via WMMA;
// enc = mean(h2[0:11]) @ w3 + b3 (VALU, f32).
// ---------------------------------------------------------------------------
__global__ __launch_bounds__(128) void k_setenc(
    const float* __restrict__ coords,
    const float* __restrict__ w1, const float* __restrict__ b1,
    const h16* __restrict__ w2T, const float* __restrict__ b2,
    const float* __restrict__ w3, const float* __restrict__ b3,
    h16* __restrict__ enc) {
  __shared__ h16   h1[16 * 128];
  __shared__ float h2[16 * 128];
  __shared__ float mv[128];
  const int tid  = threadIdx.x;
  const int wave = tid >> 5, lane = tid & 31;
  const int l15  = lane & 15, lhi = (lane & 16) ? 8 : 0;
  const int team = blockIdx.x & 1;
  const int bt   = blockIdx.x >> 1;
  const int b = bt / 200, t = bt % 200;

  for (int i = tid; i < 16 * 128; i += 128) {
    const int p = i >> 7, j = i & 127;
    float v = 0.0f;
    if (p < 11) {
      const float* cp = coords + ((size_t)b * 200 + t) * 88 + (team * 11 + p) * 4;
      const float px = cp[0] * (1.0f / 105.0f), py = cp[1] * (1.0f / 68.0f);
      v = fmaxf(0.0f, px * w1[j] + py * w1[128 + j] + b1[j]);
    }
    h1[i] = (h16)v;
  }
  __syncthreads();

  const v8f vz = {0, 0, 0, 0, 0, 0, 0, 0};
  v8f a0 = vz, a1 = vz;
  const int n0 = wave * 32, n1 = wave * 32 + 16;
#pragma unroll
  for (int kk = 0; kk < 4; ++kk) {
    const int k0 = kk * 32;
    const v16h af  = load_frag(h1 + k0, 128);
    const v16h bf0 = load_frag(w2T + (size_t)n0 * 128 + k0, 128);
    const v16h bf1 = load_frag(w2T + (size_t)n1 * 128 + k0, 128);
    a0 = wmma_f16(af, bf0, a0);
    a1 = wmma_f16(af, bf1, a1);
  }
#pragma unroll
  for (int r = 0; r < 8; ++r) {
    const int m = r + lhi;
    h2[m * 128 + n0 + l15] = fmaxf(0.0f, a0[r] + b2[n0 + l15]);
    h2[m * 128 + n1 + l15] = fmaxf(0.0f, a1[r] + b2[n1 + l15]);
  }
  __syncthreads();
  if (tid < 128) {
    float s = 0.0f;
    for (int p = 0; p < 11; ++p) s += h2[p * 128 + tid];
    mv[tid] = s * (1.0f / 11.0f);
  }
  __syncthreads();
  for (int j = tid; j < 256; j += 128) {
    float s = b3[j];
    for (int k = 0; k < 128; ++k) s += mv[k] * w3[k * 256 + j];
    enc[((size_t)t * 64 + b) * 512 + team * 256 + j] = (h16)s;
  }
}

// ---------------------------------------------------------------------------
// Traj stacked LSTM, one full step, all 4 layers fused.
// grid 22 workgroups x 256 thr; each owns 64 of the 1408 (b*NPL) rows, so the
// whole layer chain for those rows is local (no cross-WG K reads).
// Gates (512) per layer via WMMA in 2 passes: pass0 (i,g) stashed f16 in LDS,
// pass1 (f,o) in regs -> activation.  Also computes pso = glu(lin_single).
// ---------------------------------------------------------------------------
__global__ __launch_bounds__(256) void k_traj_step(
    int t, const float* __restrict__ coords,
    const float* __restrict__ wih0,   // (512,2) f32
    const h16* __restrict__ wihL,     // (3,512,128) f16
    const h16* __restrict__ whh,      // (4,512,128) f16
    const float* __restrict__ bsum,   // (4,512)
    h16* __restrict__ hbuf,           // (4,2,1408,128) f16 (step-parity)
    float* __restrict__ cst,          // (4,1408,128) f32
    const float* __restrict__ lsw, const float* __restrict__ lsb,
    float* __restrict__ pso_buf,      // (64,44)
    float* __restrict__ out)          // d_out (64,200,46)
{
  __shared__ h16   a_in[64 * 128];    // current-step layer input (f16)
  __shared__ h16   stash[64 * 256];   // i | g gate stash
  __shared__ float xin[64 * 2];
  __shared__ float po[64 * 4];

  const int tid  = threadIdx.x;
  const int wave = tid >> 5, lane = tid & 31;
  const int l15  = lane & 15, lhi = (lane & 16) ? 8 : 0;
  const int r0   = blockIdx.x * 64;
  const int pR   = t & 1, pW = pR ^ 1;

  if (tid < 64) {
    const int gr = r0 + tid;
    const int b = gr / 22, p = gr % 22;
    const float* cp = coords + ((size_t)b * 200 + t) * 88 + p * 4;
    xin[tid * 2 + 0] = cp[0] * (1.0f / 105.0f);
    xin[tid * 2 + 1] = cp[1] * (1.0f / 68.0f);
  }
  __syncthreads();

  for (int l = 0; l < 4; ++l) {
    const h16* hprev = hbuf + (((size_t)l * 2 + pR) * 1408 + r0) * 128;
    const h16* Wh = whh + (size_t)l * 512 * 128;
    const h16* Wi = (l > 0) ? (wihL + (size_t)(l - 1) * 512 * 128) : nullptr;

    for (int pass = 0; pass < 2; ++pass) {
      const int gA = pass ? 128 : 0;    // i / f block
      const int gB = pass ? 384 : 256;  // g / o block
      const int nA = gA + wave * 16;
      const int nB = gB + wave * 16;
      const v8f vz = {0, 0, 0, 0, 0, 0, 0, 0};
      v8f acc[4][2];
#pragma unroll
      for (int mt = 0; mt < 4; ++mt) { acc[mt][0] = vz; acc[mt][1] = vz; }

      // recurrent term: h_l(t-1) @ Whh^T, K = 128
#pragma unroll
      for (int kk = 0; kk < 4; ++kk) {
        const int k0 = kk * 32;
        const v16h b0 = load_frag(Wh + (size_t)nA * 128 + k0, 128);
        const v16h b1 = load_frag(Wh + (size_t)nB * 128 + k0, 128);
#pragma unroll
        for (int mt = 0; mt < 4; ++mt) {
          const v16h a = load_frag(hprev + (size_t)mt * 16 * 128 + k0, 128);
          acc[mt][0] = wmma_f16(a, b0, acc[mt][0]);
          acc[mt][1] = wmma_f16(a, b1, acc[mt][1]);
        }
      }
      if (l > 0) {
        // input term: h_{l-1}(t) @ WihL^T, K = 128 (from LDS)
#pragma unroll
        for (int kk = 0; kk < 4; ++kk) {
          const int k0 = kk * 32;
          const v16h b0 = load_frag(Wi + (size_t)nA * 128 + k0, 128);
          const v16h b1 = load_frag(Wi + (size_t)nB * 128 + k0, 128);
#pragma unroll
          for (int mt = 0; mt < 4; ++mt) {
            const v16h a = load_frag(a_in + mt * 16 * 128 + k0, 128);
            acc[mt][0] = wmma_f16(a, b0, acc[mt][0]);
            acc[mt][1] = wmma_f16(a, b1, acc[mt][1]);
          }
        }
      } else {
        // layer 0 input (K=2): VALU
        const float wA0 = wih0[(nA + l15) * 2 + 0], wA1 = wih0[(nA + l15) * 2 + 1];
        const float wB0 = wih0[(nB + l15) * 2 + 0], wB1 = wih0[(nB + l15) * 2 + 1];
#pragma unroll
        for (int mt = 0; mt < 4; ++mt)
#pragma unroll
          for (int r = 0; r < 8; ++r) {
            const int m = mt * 16 + r + lhi;
            const float x0 = xin[m * 2], x1 = xin[m * 2 + 1];
            acc[mt][0][r] += x0 * wA0 + x1 * wA1;
            acc[mt][1][r] += x0 * wB0 + x1 * wB1;
          }
      }
      const float bA = bsum[l * 512 + nA + l15];
      const float bB = bsum[l * 512 + nB + l15];
#pragma unroll
      for (int mt = 0; mt < 4; ++mt)
#pragma unroll
        for (int r = 0; r < 8; ++r) { acc[mt][0][r] += bA; acc[mt][1][r] += bB; }

      if (pass == 0) {
        const int ci = wave * 16 + l15;
#pragma unroll
        for (int mt = 0; mt < 4; ++mt)
#pragma unroll
          for (int r = 0; r < 8; ++r) {
            const int m = mt * 16 + r + lhi;
            stash[m * 256 + ci]       = (h16)acc[mt][0][r];  // i
            stash[m * 256 + 128 + ci] = (h16)acc[mt][1][r];  // g
          }
        __syncthreads();
      } else {
        __syncthreads();  // all a_in reads + stash writes done
        const int ncl = wave * 16 + l15;
#pragma unroll
        for (int mt = 0; mt < 4; ++mt)
#pragma unroll
          for (int r = 0; r < 8; ++r) {
            const int m  = mt * 16 + r + lhi;
            const float iv = (float)stash[m * 256 + ncl];
            const float gv = (float)stash[m * 256 + 128 + ncl];
            const float fv = acc[mt][0][r];
            const float ov = acc[mt][1][r];
            const int gr = r0 + m;
            const size_t ci = ((size_t)l * 1408 + gr) * 128 + ncl;
            const float co = cst[ci];
            const float cn = sigf(fv) * co + sigf(iv) * tanhf(gv);
            const float hn = sigf(ov) * tanhf(cn);
            cst[ci] = cn;
            a_in[m * 128 + ncl] = (h16)hn;  // next layer's input
            hbuf[(((size_t)l * 2 + pW) * 1408 + gr) * 128 + ncl] = (h16)hn;
          }
        __syncthreads();
      }
    }
  }

  // pso = glu(h3 @ lin_single_w^T + b)  (4 outputs -> 2 per row)
  {
    const int m = tid >> 2, j = tid & 3;
    float s = lsb[j];
    for (int k = 0; k < 128; ++k) s += (float)a_in[m * 128 + k] * lsw[j * 128 + k];
    po[m * 4 + j] = s;
  }
  __syncthreads();
  if (tid < 128) {
    const int m = tid >> 1, jj = tid & 1;
    const float v = po[m * 4 + jj] * sigf(po[m * 4 + jj + 2]);
    const int gr = r0 + m;
    const int b = gr / 22, p = gr % 22;
    pso_buf[b * 44 + 2 * p + jj] = v;
    out[((size_t)b * 200 + t) * 46 + 2 * p + jj] = v * (jj ? 68.0f : 105.0f);
  }
}

// ---------------------------------------------------------------------------
// rnn_in assembly: [ptrace(44) | enc(512) | pso(44) | sorted dists(22) |
//                   step_out(2) | zero pad(16)]  -> f16 (64 x 640)
// ---------------------------------------------------------------------------
__global__ __launch_bounds__(256) void k_assemble(
    int t, const float* __restrict__ coords, const h16* __restrict__ enc,
    const float* __restrict__ pso_buf, const float* __restrict__ sout,
    h16* __restrict__ rnn) {
  const int tid = threadIdx.x;
  for (int i = tid; i < 64 * 22; i += 256) {
    const int b = i / 22, p = i % 22;
    const float* cp = coords + ((size_t)b * 200 + t) * 88 + p * 4;
    rnn[b * 640 + 2 * p + 0] = (h16)(cp[0] * (1.0f / 105.0f));
    rnn[b * 640 + 2 * p + 1] = (h16)(cp[1] * (1.0f / 68.0f));
  }
  for (int i = tid; i < 64 * 512; i += 256) {
    const int b = i >> 9, k = i & 511;
    rnn[b * 640 + 44 + k] = enc[((size_t)t * 64 + b) * 512 + k];
  }
  for (int i = tid; i < 64 * 44; i += 256) {
    const int b = i / 44, j = i % 44;
    rnn[b * 640 + 556 + j] = (h16)pso_buf[b * 44 + j];
  }
  for (int i = tid; i < 64 * 18; i += 256) {
    const int b = i / 18, j = i % 18;
    rnn[b * 640 + 622 + j] = (j < 2) ? (h16)sout[b * 2 + j] : (h16)0.0f;
  }
  if (tid < 64) {
    const int b = tid;
    const float px = sout[b * 2 + 0], py = sout[b * 2 + 1];
    float d[22];
    for (int p = 0; p < 22; ++p) {
      const float* cp = coords + ((size_t)b * 200 + t) * 88 + p * 4;
      const float dx = cp[0] * (1.0f / 105.0f) - px;
      const float dy = cp[1] * (1.0f / 68.0f) - py;
      d[p] = sqrtf(dx * dx + dy * dy);
    }
    for (int i = 1; i < 22; ++i) {  // insertion sort ascending
      const float v = d[i];
      int j = i - 1;
      while (j >= 0 && d[j] > v) { d[j + 1] = d[j]; --j; }
      d[j + 1] = v;
    }
    for (int i = 0; i < 22; ++i) rnn[b * 640 + 600 + i] = (h16)d[i];
  }
}

// ---------------------------------------------------------------------------
// One LSTM layer of the ball & pos stacks (blockIdx.y selects the stack).
// grid.x = 8 hidden slices of 64 cols; each workgroup computes i,f,g,o for
// its slice (gate rows slice+{0,512,1024,1536}) so activation is local.
// Pass0 -> (i,g) stash f16; pass1 -> (f stash, o regs) -> c/h update.
// ---------------------------------------------------------------------------
__global__ __launch_bounds__(256) void k_lstm_layer(
    int t, int l,
    const h16* __restrict__ rnn,       // (64,640)
    const h16* __restrict__ wih0_bp,   // (2,2048,640)
    const h16* __restrict__ wihL_bp,   // (2,3,2048,512)
    const h16* __restrict__ whh_bp,    // (2,4,2048,512)
    const float* __restrict__ bsum_bp, // (2,4,2048)
    h16* __restrict__ hbuf_bp,         // (2,4,2,64,512) step-parity
    float* __restrict__ c_bp)          // (2,4,64,512)
{
  __shared__ h16 stash[64 * 192];
  const int tid  = threadIdx.x;
  const int wave = tid >> 5, lane = tid & 31;
  const int l15  = lane & 15, lhi = (lane & 16) ? 8 : 0;
  const int ls   = blockIdx.y;
  const int cb   = blockIdx.x * 64;
  const int pR   = t & 1, pW = pR ^ 1;
  const int gsel = wave >> 2;
  const int ncol = (wave & 3) * 16;

  const int Ki = (l == 0) ? 640 : 512;
  const h16* A1 = (l == 0)
      ? rnn
      : hbuf_bp + (((size_t)ls * 4 + (l - 1)) * 2 + pW) * 64 * 512;
  const h16* Ah = hbuf_bp + (((size_t)ls * 4 + l) * 2 + pR) * 64 * 512;
  const h16* Wi = (l == 0)
      ? wih0_bp + (size_t)ls * 2048 * 640
      : wihL_bp + ((size_t)ls * 3 + (l - 1)) * 2048 * 512;
  const h16* Wh = whh_bp + ((size_t)ls * 4 + l) * 2048 * 512;
  const float* bs = bsum_bp + ((size_t)ls * 4 + l) * 2048;
  float* cst = c_bp + ((size_t)ls * 4 + l) * 64 * 512;
  h16* hout = hbuf_bp + (((size_t)ls * 4 + l) * 2 + pW) * 64 * 512;

  for (int pass = 0; pass < 2; ++pass) {
    const int grow = (pass ? (gsel ? 3 * 512 : 512) : (gsel ? 2 * 512 : 0)) + cb + ncol;
    const v8f vz = {0, 0, 0, 0, 0, 0, 0, 0};
    v8f acc[4];
#pragma unroll
    for (int mt = 0; mt < 4; ++mt) acc[mt] = vz;

    for (int kk = 0; kk < Ki / 32; ++kk) {   // input term
      const int k0 = kk * 32;
      const v16h bf = load_frag(Wi + (size_t)grow * Ki + k0, Ki);
#pragma unroll
      for (int mt = 0; mt < 4; ++mt) {
        const v16h af = load_frag(A1 + (size_t)mt * 16 * Ki + k0, Ki);
        acc[mt] = wmma_f16(af, bf, acc[mt]);
      }
    }
#pragma unroll 4
    for (int kk = 0; kk < 16; ++kk) {        // recurrent term (K=512)
      const int k0 = kk * 32;
      const v16h bf = load_frag(Wh + (size_t)grow * 512 + k0, 512);
#pragma unroll
      for (int mt = 0; mt < 4; ++mt) {
        const v16h af = load_frag(Ah + (size_t)mt * 16 * 512 + k0, 512);
        acc[mt] = wmma_f16(af, bf, acc[mt]);
      }
    }
    const float bv = bs[grow + l15];
#pragma unroll
    for (int mt = 0; mt < 4; ++mt)
#pragma unroll
      for (int r = 0; r < 8; ++r) acc[mt][r] += bv;

    if (pass == 0) {
      const int sc = gsel * 64 + ncol + l15;  // i -> 0..63, g -> 64..127
#pragma unroll
      for (int mt = 0; mt < 4; ++mt)
#pragma unroll
        for (int r = 0; r < 8; ++r)
          stash[(mt * 16 + r + lhi) * 192 + sc] = (h16)acc[mt][r];
      __syncthreads();
    } else {
      if (gsel == 0) {                        // f -> 128..191
#pragma unroll
        for (int mt = 0; mt < 4; ++mt)
#pragma unroll
          for (int r = 0; r < 8; ++r)
            stash[(mt * 16 + r + lhi) * 192 + 128 + ncol + l15] = (h16)acc[mt][r];
      }
      __syncthreads();
      if (gsel == 1) {                        // o in regs -> combine
#pragma unroll
        for (int mt = 0; mt < 4; ++mt)
#pragma unroll
          for (int r = 0; r < 8; ++r) {
            const int m   = mt * 16 + r + lhi;
            const int ncl = ncol + l15;
            const float iv = (float)stash[m * 192 + ncl];
            const float gv = (float)stash[m * 192 + 64 + ncl];
            const float fv = (float)stash[m * 192 + 128 + ncl];
            const float ov = acc[mt][r];
            const size_t ci = (size_t)m * 512 + cb + ncl;
            const float co = cst[ci];
            const float cn = sigf(fv) * co + sigf(iv) * tanhf(gv);
            const float hn = sigf(ov) * tanhf(cn);
            cst[ci]  = cn;
            hout[ci] = (h16)hn;
          }
      }
    }
  }
}

// ---------------------------------------------------------------------------
// GLU heads: new_step = glu(h_ball3 @ lin_w^T + b); pos = glu(h_pos3 @ ...).
// ---------------------------------------------------------------------------
__global__ __launch_bounds__(256) void k_output(
    int t, const h16* __restrict__ hball, const h16* __restrict__ hpos,
    const float* __restrict__ lw, const float* __restrict__ lb,
    const float* __restrict__ plw, const float* __restrict__ plb,
    float* __restrict__ sout, float* __restrict__ out) {
  __shared__ float raw[64 * 10];
  const int tid = threadIdx.x;
  for (int i = tid; i < 640; i += 256) {
    const int b = i / 10, j = i % 10;
    float s;
    if (j < 4) {
      s = lb[j];
      for (int k = 0; k < 512; ++k) s += (float)hball[b * 512 + k] * lw[j * 512 + k];
    } else {
      const int jj = j - 4;
      s = plb[jj];
      for (int k = 0; k < 512; ++k) s += (float)hpos[b * 512 + k] * plw[jj * 512 + k];
    }
    raw[b * 10 + j] = s;
  }
  __syncthreads();
  for (int i = tid; i < 64 * 5; i += 256) {
    const int b = i / 5, j = i % 5;
    if (j < 2) {
      const float v = raw[b * 10 + j] * sigf(raw[b * 10 + j + 2]);
      sout[b * 2 + j] = v;
      out[((size_t)b * 200 + t) * 46 + 44 + j] = v * (j ? 68.0f : 105.0f);
    } else {
      const int jj = j - 2;
      const float v = raw[b * 10 + 4 + jj] * sigf(raw[b * 10 + 4 + jj + 3]);
      out[(size_t)64 * 200 * 46 + ((size_t)b * 200 + t) * 3 + jj] = v;
    }
  }
}

// ---------------------------------------------------------------------------
// Prep kernels: f32->f16 (opt. K padding / transpose), bias sums, zeroing.
// ---------------------------------------------------------------------------
__global__ void k_cvt(const float* __restrict__ s, h16* __restrict__ d, long n) {
  for (long i = blockIdx.x * (long)blockDim.x + threadIdx.x; i < n;
       i += (long)gridDim.x * blockDim.x)
    d[i] = (h16)s[i];
}
__global__ void k_cvt_pad(const float* __restrict__ s, h16* __restrict__ d,
                          int rows, int kin, int kout) {
  const long n = (long)rows * kout;
  for (long i = blockIdx.x * (long)blockDim.x + threadIdx.x; i < n;
       i += (long)gridDim.x * blockDim.x) {
    const int r = (int)(i / kout), k = (int)(i % kout);
    d[i] = (k < kin) ? (h16)s[(size_t)r * kin + k] : (h16)0.0f;
  }
}
__global__ void k_tr(const float* __restrict__ s, h16* __restrict__ d, int n128) {
  const long n = (long)n128 * n128;
  for (long i = blockIdx.x * (long)blockDim.x + threadIdx.x; i < n;
       i += (long)gridDim.x * blockDim.x) {
    const int nn = (int)(i / n128), k = (int)(i % n128);
    d[i] = (h16)s[(size_t)k * n128 + nn];
  }
}
__global__ void k_bsum(const float* __restrict__ a, const float* __restrict__ b,
                       float* __restrict__ d, long n) {
  for (long i = blockIdx.x * (long)blockDim.x + threadIdx.x; i < n;
       i += (long)gridDim.x * blockDim.x)
    d[i] = a[i] + b[i];
}
__global__ void k_zero(uint32_t* __restrict__ p, long n) {
  for (long i = blockIdx.x * (long)blockDim.x + threadIdx.x; i < n;
       i += (long)gridDim.x * blockDim.x)
    p[i] = 0u;
}

// ---------------------------------------------------------------------------
extern "C" void kernel_launch(void* const* d_in, const int* in_sizes, int n_in,
                              void* d_out, int out_size, void* d_ws, size_t ws_size,
                              hipStream_t stream) {
  (void)in_sizes; (void)n_in; (void)out_size; (void)ws_size;
  const float* coords = (const float*)d_in[1];
  const float* stf_w1 = (const float*)d_in[3];
  const float* stf_b1 = (const float*)d_in[4];
  const float* stf_w2 = (const float*)d_in[5];
  const float* stf_b2 = (const float*)d_in[6];
  const float* stf_w3 = (const float*)d_in[7];
  const float* stf_b3 = (const float*)d_in[8];
  const float* t_wih0 = (const float*)d_in[9];
  const float* t_wihL = (const float*)d_in[10];
  const float* t_whh  = (const float*)d_in[11];
  const float* t_bih  = (const float*)d_in[12];
  const float* t_bhh  = (const float*)d_in[13];
  const float* b_wih0 = (const float*)d_in[14];
  const float* b_wihL = (const float*)d_in[15];
  const float* b_whh  = (const float*)d_in[16];
  const float* b_bih  = (const float*)d_in[17];
  const float* b_bhh  = (const float*)d_in[18];
  const float* p_wih0 = (const float*)d_in[19];
  const float* p_wihL = (const float*)d_in[20];
  const float* p_whh  = (const float*)d_in[21];
  const float* p_bih  = (const float*)d_in[22];
  const float* p_bhh  = (const float*)d_in[23];
  const float* lin_w  = (const float*)d_in[24];
  const float* lin_b  = (const float*)d_in[25];
  const float* lsw    = (const float*)d_in[26];
  const float* lsb    = (const float*)d_in[27];
  const float* plw    = (const float*)d_in[28];
  const float* plb    = (const float*)d_in[29];
  float* out = (float*)d_out;

  char* ws = (char*)d_ws;
  size_t off = 0;
  auto take = [&](size_t bytes) -> void* {
    void* p = ws + off;
    off = (off + bytes + 255) & ~(size_t)255;
    return p;
  };
  h16*   wih0_bp = (h16*)take(2ull * 2048 * 640 * 2);
  h16*   wihL_bp = (h16*)take(2ull * 3 * 2048 * 512 * 2);
  h16*   whh_bp  = (h16*)take(2ull * 4 * 2048 * 512 * 2);
  h16*   wihL_t  = (h16*)take(3ull * 512 * 128 * 2);
  h16*   whh_t   = (h16*)take(4ull * 512 * 128 * 2);
  h16*   w2T     = (h16*)take(128ull * 128 * 2);
  float* bsum_bp = (float*)take(2ull * 4 * 2048 * 4);
  float* bsum_t  = (float*)take(4ull * 512 * 4);
  h16*   enc     = (h16*)take(200ull * 64 * 512 * 2);
  h16*   rnn     = (h16*)take(64ull * 640 * 2);
  h16*   hbuf_bp = (h16*)take(2ull * 4 * 2 * 64 * 512 * 2);
  float* c_bp    = (float*)take(2ull * 4 * 64 * 512 * 4);
  h16*   hbuf_t  = (h16*)take(4ull * 2 * 1408 * 128 * 2);
  float* c_t     = (float*)take(4ull * 1408 * 128 * 4);
  float* pso     = (float*)take(64ull * 44 * 4);
  float* sout    = (float*)take(64ull * 2 * 4);

  const dim3 cg(1024), cb(256);
  // f16 weight banks (ball = bank0, pos = bank1), K padded 624->640
  k_cvt_pad<<<cg, cb, 0, stream>>>(b_wih0, wih0_bp, 2048, 624, 640);
  k_cvt_pad<<<cg, cb, 0, stream>>>(p_wih0, wih0_bp + 2048ull * 640, 2048, 624, 640);
  k_cvt<<<cg, cb, 0, stream>>>(b_wihL, wihL_bp, 3L * 2048 * 512);
  k_cvt<<<cg, cb, 0, stream>>>(p_wihL, wihL_bp + 3ull * 2048 * 512, 3L * 2048 * 512);
  k_cvt<<<cg, cb, 0, stream>>>(b_whh, whh_bp, 4L * 2048 * 512);
  k_cvt<<<cg, cb, 0, stream>>>(p_whh, whh_bp + 4ull * 2048 * 512, 4L * 2048 * 512);
  k_cvt<<<cg, cb, 0, stream>>>(t_wihL, wihL_t, 3L * 512 * 128);
  k_cvt<<<cg, cb, 0, stream>>>(t_whh, whh_t, 4L * 512 * 128);
  k_tr<<<cg, cb, 0, stream>>>(stf_w2, w2T, 128);
  k_bsum<<<cg, cb, 0, stream>>>(b_bih, b_bhh, bsum_bp, 4L * 2048);
  k_bsum<<<cg, cb, 0, stream>>>(p_bih, p_bhh, bsum_bp + 4ull * 2048, 4L * 2048);
  k_bsum<<<cg, cb, 0, stream>>>(t_bih, t_bhh, bsum_t, 4L * 512);
  // zero recurrent state (carry0) every call - deterministic
  k_zero<<<cg, cb, 0, stream>>>((uint32_t*)hbuf_bp, (long)(2ull * 4 * 2 * 64 * 512 * 2 / 4));
  k_zero<<<cg, cb, 0, stream>>>((uint32_t*)c_bp, (long)(2ull * 4 * 64 * 512));
  k_zero<<<cg, cb, 0, stream>>>((uint32_t*)hbuf_t, (long)(4ull * 2 * 1408 * 128 * 2 / 4));
  k_zero<<<cg, cb, 0, stream>>>((uint32_t*)c_t, (long)(4ull * 1408 * 128));
  k_zero<<<cg, cb, 0, stream>>>((uint32_t*)sout, 32L);

  // set encoder for all (b,t,team) - fully parallel, WMMA
  k_setenc<<<dim3(64 * 200 * 2), dim3(128), 0, stream>>>(
      coords, stf_w1, stf_b1, w2T, stf_b2, stf_w3, stf_b3, enc);

  // sequential recurrence over T
  for (int t = 0; t < 200; ++t) {
    const int pW = (t & 1) ^ 1;
    k_traj_step<<<dim3(22), dim3(256), 0, stream>>>(
        t, coords, t_wih0, wihL_t, whh_t, bsum_t, hbuf_t, c_t, lsw, lsb, pso, out);
    k_assemble<<<dim3(1), dim3(256), 0, stream>>>(t, coords, enc, pso, sout, rnn);
    for (int l = 0; l < 4; ++l)
      k_lstm_layer<<<dim3(8, 2), dim3(256), 0, stream>>>(
          t, l, rnn, wih0_bp, wihL_bp, whh_bp, bsum_bp, hbuf_bp, c_bp);
    const h16* hball = hbuf_bp + (((size_t)0 * 4 + 3) * 2 + pW) * 64 * 512;
    const h16* hpos  = hbuf_bp + (((size_t)1 * 4 + 3) * 2 + pW) * 64 * 512;
    k_output<<<dim3(1), dim3(256), 0, stream>>>(
        t, hball, hpos, lin_w, lin_b, plw, plb, sout, out);
  }
}
